// myRNN_53644141527229
// MI455X (gfx1250) — matrix operand order repaired and checked by reading, not compiled, not run on hardware
//
#include <hip/hip_runtime.h>
#include <hip/hip_bf16.h>
#include <math.h>

typedef __attribute__((ext_vector_type(16))) _Float16 v16h;
typedef __attribute__((ext_vector_type(8)))  _Float16 v8h;
typedef __attribute__((ext_vector_type(8)))  float    v8f;

#define RNN_N 512
#define RNN_T 512
#define RNN_B 256
#define DECAY 0.1f

// ---------------------------------------------------------------------------
// Kernel 0: pack W (f32, [n][k], row-major) into f16 B-fragment order so that
// each lane's 16x16x32 WMMA B-fragment is one contiguous 32-byte chunk.
// Flat layout: (((ntile*16 + ktile)*32 + lane)*16 + j)
//   n = ntile*16 + (lane & 15)
//   k = ktile*32 + (lane >> 4)*8 + (j & 7) + (j >> 3)*16
//   value = B[k][n] = W[n][k]
// ---------------------------------------------------------------------------
__global__ void pack_w_kernel(const float* __restrict__ W,
                              _Float16* __restrict__ Wpk) {
    int idx  = blockIdx.x * blockDim.x + threadIdx.x;   // 0 .. N*N-1
    int j    = idx & 15;
    int lane = (idx >> 4) & 31;
    int kt   = (idx >> 9) & 15;
    int nt   = idx >> 13;
    int n = nt * 16 + (lane & 15);
    int k = kt * 32 + (lane >> 4) * 8 + (j & 7) + (j >> 3) * 16;
    Wpk[idx] = (_Float16)W[n * RNN_N + k];
}

// ---------------------------------------------------------------------------
// Kernel 1: the RNN scan.
// Grid: 16 workgroups (16 batch rows each), 1024 threads = 32 wave32s.
// Wave w owns N-tile [16w, 16w+16). Hidden state h lives in registers in the
// WMMA C-layout (8 f32/lane). tanh(h) is staged in LDS as f16 [16][512]
// (m-major) so A-fragments are two contiguous ds_load_b128 per K-tile.
// B fragments are re-streamed from L2 each step (opaque offset defeats LICM
// so the compiler cannot hoist 128 VGPRs of W and spill to scratch).
// ---------------------------------------------------------------------------
__global__ void __launch_bounds__(1024)
rnn_scan_kernel(const float* __restrict__ h0,
                const float* __restrict__ X,      // [B,T,N]
                const _Float16* __restrict__ Wpk, // packed fragments
                float* __restrict__ Hout) {       // [B,T,N] = tanh(h_t)
    __shared__ _Float16 sTH[16 * RNN_N];          // 16 KB

    const int lane     = threadIdx.x & 31;
    const int wave     = threadIdx.x >> 5;        // 0..31  -> N-tile
    const int laneHalf = lane >> 4;               // 0/1
    const int lmod     = lane & 15;
    const int b0       = blockIdx.x * 16;         // first batch row of tile
    const int nCol     = wave * 16 + lmod;        // this lane's column n

    // hidden state in C-layout: element r -> (m = r + 8*laneHalf, n = nCol)
    float h[8];
    const float hinit = h0[nCol];
#pragma unroll
    for (int r = 0; r < 8; ++r) h[r] = hinit;

    // stage tanh(h0) into LDS (same value for every m, since h0 broadcast)
    for (int i = threadIdx.x; i < 16 * RNN_N; i += 1024) {
        int k = i & (RNN_N - 1);
        sTH[i] = (_Float16)tanhf(h0[k]);
    }
    __syncthreads();

    // B fragments: wave-private contiguous 16 KB region, re-streamed from L2
    const _Float16* wbase = Wpk + (size_t)wave * (16 * 512) + lane * 16;
    // A fragment LDS base (in halfs): row m = lmod, K chunk at 8*laneHalf
    const int aRow = lmod * RNN_N + 8 * laneHalf;

    const size_t bt_stride = (size_t)RNN_T * RNN_N;

    for (int t = 0; t < RNN_T; ++t) {
        // opaque zero, re-defined every iteration: B loads cannot be hoisted
        int zoff = 0;
        asm volatile("" : "+s"(zoff));
        const _Float16* wb = wbase + zoff;

        // prefetch x_t into registers; overlaps with the WMMA chain below
        float xv[8];
#pragma unroll
        for (int r = 0; r < 8; ++r) {
            int m = r + 8 * laneHalf;
            xv[r] = X[(size_t)(b0 + m) * bt_stride + (size_t)t * RNN_N + nCol];
        }

        // --- K loop, software-pipelined by one stage ---
        v16h acur, bcur;
        *(v8h*)&acur       = *(const v8h*)&sTH[aRow];
        *((v8h*)&acur + 1) = *(const v8h*)&sTH[aRow + 16];
        bcur = *(const v16h*)(wb);

        v8f c = {};
#pragma unroll
        for (int kt = 0; kt < 16; ++kt) {
            const int ktn = (kt + 1) & 15;   // last stage reloads kt=0 (dead)
            v16h anxt, bnxt;
            *(v8h*)&anxt       = *(const v8h*)&sTH[aRow + ktn * 32];
            *((v8h*)&anxt + 1) = *(const v8h*)&sTH[aRow + ktn * 32 + 16];
            bnxt = *(const v16h*)(wb + ktn * 512);

            c = __builtin_amdgcn_wmma_f32_16x16x32_f16(
                    false, acur, false, bcur, (short)0, c, false, false);
            acur = anxt;
            bcur = bnxt;
        }
        __syncthreads();   // everyone done reading sTH for this step

        // h_new = 0.9 h + 0.1 (C + x);  emit tanh(h_new) to LDS + global
#pragma unroll
        for (int r = 0; r < 8; ++r) {
            int m = r + 8 * laneHalf;
            float hn = (1.0f - DECAY) * h[r] + DECAY * (c[r] + xv[r]);
            h[r] = hn;
            float th = tanhf(hn);
            sTH[m * RNN_N + nCol] = (_Float16)th;
            Hout[(size_t)(b0 + m) * bt_stride + (size_t)t * RNN_N + nCol] = th;
        }
        __syncthreads();   // sTH fully updated before next step's reads
    }
}

// ---------------------------------------------------------------------------
// Kernel 2: geometry = hidden @ Gw^T + Gb, one wave per (b,t).
// Lane l accumulates n = l, l+32, ... (coalesced), then wave32 shfl-reduce.
// ---------------------------------------------------------------------------
__global__ void geometry_kernel(const float* __restrict__ H,   // [B,T,N]
                                const float* __restrict__ Gw,  // [2,N]
                                const float* __restrict__ Gb,  // [2]
                                float* __restrict__ Geo) {     // [B,T,2]
    int gwave = (blockIdx.x * blockDim.x + threadIdx.x) >> 5;  // (b,t) index
    int lane  = threadIdx.x & 31;
    const float* hrow = H + (size_t)gwave * RNN_N;

    float s0 = 0.0f, s1 = 0.0f;
#pragma unroll 4
    for (int k = lane; k < RNN_N; k += 32) {
        float v = hrow[k];
        s0 += v * Gw[k];
        s1 += v * Gw[RNN_N + k];
    }
#pragma unroll
    for (int off = 16; off > 0; off >>= 1) {
        s0 += __shfl_down(s0, off, 32);
        s1 += __shfl_down(s1, off, 32);
    }
    if (lane == 0) {
        Geo[2 * (size_t)gwave]     = s0 + Gb[0];
        Geo[2 * (size_t)gwave + 1] = s1 + Gb[1];
    }
}

// ---------------------------------------------------------------------------
extern "C" void kernel_launch(void* const* d_in, const int* in_sizes, int n_in,
                              void* d_out, int out_size, void* d_ws, size_t ws_size,
                              hipStream_t stream) {
    const float* h0 = (const float*)d_in[0];   // [N]
    const float* X  = (const float*)d_in[1];   // [B,T,N]
    const float* W  = (const float*)d_in[2];   // [N,N]
    const float* Gw = (const float*)d_in[3];   // [2,N]
    const float* Gb = (const float*)d_in[4];   // [2]

    float* Hout = (float*)d_out;                                   // [B,T,N]
    float* Geo  = Hout + (size_t)RNN_B * RNN_T * RNN_N;            // [B,T,2]
    _Float16* Wpk = (_Float16*)d_ws;                               // 512 KB

    // 1) pack W into f16 WMMA-B fragment order
    pack_w_kernel<<<(RNN_N * RNN_N) / 256, 256, 0, stream>>>(W, Wpk);

    // 2) the scan: 16 WGs x 32 waves
    rnn_scan_kernel<<<16, 1024, 0, stream>>>(h0, X, Wpk, Hout);

    // 3) geometry projection: one wave per (b,t)
    geometry_kernel<<<(RNN_B * RNN_T) / 8, 256, 0, stream>>>(Hout, Gw, Gb, Geo);
}
